// SwitchFeedForward_57578331570208
// MI455X (gfx1250) — compile-verified, matrix operand order checked by reading, not run on hardware
//
#include <hip/hip_runtime.h>
#include <hip/hip_bf16.h>

typedef __attribute__((ext_vector_type(16))) __bf16 v16bf;
typedef __attribute__((ext_vector_type(8)))  float  v8f;
typedef unsigned short u16;

#define T_TOK    16384
#define DIM      512
#define FFDIM    2048
#define NEXP     8
#define AP       520    // sA / sB1 row pitch in ushorts (1040B, 16B-aligned, bank-spread)
#define HP       72     // sH / sB2 row pitch in ushorts (144B, 16B-aligned)
#define MAXTILES 264

__device__ __forceinline__ u16 f32_to_bf16(float f){
  unsigned int u = __float_as_uint(f);
  u += 0x7FFFu + ((u >> 16) & 1u);         // round-to-nearest-even
  return (u16)(u >> 16);
}

union FragU { uint4 q[2]; v16bf v; };

// bf16 A/B fragment (16x32 / 32x16): lane L -> row (rowBase + L%16);
// K elements (kb + 8*(L/16)) .. +7  and  +16 .. +23   => two b128 LDS loads.
__device__ __forceinline__ v16bf frag_load(const u16* base, int rowBase, int pitch, int kb, int lane){
  int idx  = lane & 15;
  int half = lane >> 4;
  const u16* p = base + (rowBase + idx) * pitch + kb + half * 8;
  FragU u;
  u.q[0] = *(const uint4*)(p);
  u.q[1] = *(const uint4*)(p + 16);
  return u.v;
}

// Async global->LDS copy, 16B per enabled lane; tracked by ASYNCcnt.
// GVS addressing: mem = SADDR(64) + VADDR(32) ; VDST = LDS byte address.
__device__ __forceinline__ void async_ld128(void* ldsPtr, const void* gbase, unsigned byteOff){
  unsigned lds = (unsigned)(size_t)ldsPtr;     // low 32 bits of flat LDS addr = LDS offset
  asm volatile("global_load_async_to_lds_b128 %0, %1, %2"
               :: "v"(lds), "v"(byteOff), "s"(gbase)
               : "memory");
}
#define WAIT_ASYNC_16() asm volatile("s_wait_asynccnt 16" ::: "memory")

// ---------------- weight transpose + f32->bf16 ----------------
// in: [nmat][R][C] f32, out: [nmat][C][R] bf16
__global__ void transpose_bf16_kernel(const float* __restrict__ in, u16* __restrict__ out,
                                      int R, int C){
  __shared__ float t[32][33];
  const float* I = in  + (size_t)blockIdx.z * R * C;
  u16*         O = out + (size_t)blockIdx.z * R * C;
  int c0 = blockIdx.x * 32, r0 = blockIdx.y * 32;
  int tx = threadIdx.x, ty = threadIdx.y;            // block (32,8)
  #pragma unroll
  for (int j = 0; j < 4; ++j)
    t[ty + j*8][tx] = I[(size_t)(r0 + ty + j*8) * C + c0 + tx];
  __syncthreads();
  #pragma unroll
  for (int j = 0; j < 4; ++j)
    O[(size_t)(c0 + ty + j*8) * R + r0 + tx] = f32_to_bf16(t[tx][ty + j*8]);
}

// ---------------- router: logits -> softmax pmax + argmax ----------------
__global__ __launch_bounds__(256) void router_kernel(const float* __restrict__ x,
                                                     const float* __restrict__ Wg,
                                                     const float* __restrict__ bg,
                                                     int* __restrict__ route,
                                                     float* __restrict__ pmax){
  __shared__ float sWg[DIM * NEXP];
  __shared__ float sbg[NEXP];
  int tid = threadIdx.x;
  for (int i = tid; i < DIM * NEXP; i += 256) sWg[i] = Wg[i];
  if (tid < NEXP) sbg[tid] = bg[tid];
  __syncthreads();

  int lane = tid & 31, wave = tid >> 5;
  int t = blockIdx.x * 8 + wave;                 // 8 waves -> 8 tokens / block
  const float4* xr = (const float4*)(x + (size_t)t * DIM);
  float acc[NEXP];
  #pragma unroll
  for (int e = 0; e < NEXP; ++e) acc[e] = 0.f;
  #pragma unroll
  for (int j = 0; j < 4; ++j) {
    float4 v = xr[lane * 4 + j];
    int k = lane * 16 + j * 4;
    #pragma unroll
    for (int e = 0; e < NEXP; ++e) {
      acc[e] += v.x * sWg[(k+0)*NEXP + e] + v.y * sWg[(k+1)*NEXP + e]
             +  v.z * sWg[(k+2)*NEXP + e] + v.w * sWg[(k+3)*NEXP + e];
    }
  }
  #pragma unroll
  for (int off = 16; off > 0; off >>= 1)
    #pragma unroll
    for (int e = 0; e < NEXP; ++e) acc[e] += __shfl_xor(acc[e], off);

  float m = -1e30f; int arg = 0;
  #pragma unroll
  for (int e = 0; e < NEXP; ++e) {
    float l = acc[e] + sbg[e];
    if (l > m) { m = l; arg = e; }
    acc[e] = l;
  }
  float s = 0.f;
  #pragma unroll
  for (int e = 0; e < NEXP; ++e) s += __expf(acc[e] - m);
  if (lane == 0) { route[t] = arg; pmax[t] = 1.0f / s; }
}

// ---------------- per-block expert histogram ----------------
__global__ __launch_bounds__(256) void hist_kernel(const int* __restrict__ route,
                                                   int* __restrict__ blockCounts){
  __shared__ int cnt[NEXP];
  int tid = threadIdx.x;
  if (tid < NEXP) cnt[tid] = 0;
  __syncthreads();
  atomicAdd(&cnt[route[blockIdx.x * 256 + tid]], 1);
  __syncthreads();
  if (tid < NEXP) blockCounts[blockIdx.x * NEXP + tid] = cnt[tid];
}

// ---------------- scan: offsets, stable bases, tile table ----------------
__global__ void scan_kernel(const int* __restrict__ blockCounts,
                            int* __restrict__ baseArr, int4* __restrict__ tiles){
  if (threadIdx.x != 0) return;
  const int NB = T_TOK / 256;
  int tot[NEXP], off[NEXP];
  for (int e = 0; e < NEXP; ++e) {
    int s = 0;
    for (int b = 0; b < NB; ++b) s += blockCounts[b * NEXP + e];
    tot[e] = s;
  }
  int a = 0;
  for (int e = 0; e < NEXP; ++e) { off[e] = a; a += tot[e]; }
  for (int e = 0; e < NEXP; ++e) {
    int run = off[e];
    for (int b = 0; b < NB; ++b) { baseArr[b * NEXP + e] = run; run += blockCounts[b * NEXP + e]; }
  }
  int nt = 0;
  for (int e = 0; e < NEXP; ++e) {
    int c = tot[e], p = 0;
    while (p < c) {
      int rows = c - p; if (rows > 64) rows = 64;
      tiles[nt] = make_int4(e, off[e] + p, rows, 0);
      ++nt; p += 64;
    }
  }
  for (int i = nt; i < MAXTILES; ++i) tiles[i] = make_int4(0, 0, 0, 0);
}

// ---------------- stable rank + gather tokens into expert-sorted bf16 rows ----------------
__global__ __launch_bounds__(256) void gather_kernel(const float* __restrict__ x,
                                                     const int* __restrict__ route,
                                                     const float* __restrict__ pmax,
                                                     const int* __restrict__ baseArr,
                                                     u16* __restrict__ xs_sorted){
  __shared__ int   sroute[256];
  __shared__ int   sdest[256];
  __shared__ float spmax[256];
  int tid = threadIdx.x;
  int t = blockIdx.x * 256 + tid;
  sroute[tid] = route[t];
  spmax[tid]  = pmax[t];
  __syncthreads();
  int r = sroute[tid], rank = 0;
  for (int j = 0; j < tid; ++j) rank += (sroute[j] == r);
  sdest[tid] = baseArr[blockIdx.x * NEXP + r] + rank;
  __syncthreads();

  int lane = tid & 31, wave = tid >> 5;
  for (int i = wave; i < 256; i += 8) {
    int tt = blockIdx.x * 256 + i;
    int d  = sdest[i];
    float pm = spmax[i];
    const float4* src = (const float4*)(x + (size_t)tt * DIM);
    u16* dst = xs_sorted + (size_t)d * DIM;
    #pragma unroll
    for (int j = 0; j < 4; ++j) {
      int fi = lane + 32 * j;                 // float4 index 0..127
      float4 v = src[fi];
      ushort4 o;
      o.x = f32_to_bf16(v.x * pm); o.y = f32_to_bf16(v.y * pm);
      o.z = f32_to_bf16(v.z * pm); o.w = f32_to_bf16(v.w * pm);
      *(ushort4*)(dst + fi * 4) = o;
    }
  }
}

// ---------------- fused MoE FFN: y = relu(Xs@W1+b1)@W2+b2 over 64-token tile ----------------
__global__ __launch_bounds__(256) void ffn_kernel(const u16* __restrict__ xs_sorted,
                                                  const u16* __restrict__ W1T,
                                                  const u16* __restrict__ W2T,
                                                  const float* __restrict__ b1,
                                                  const float* __restrict__ b2,
                                                  const int4* __restrict__ tiles,
                                                  float* __restrict__ out){
  __shared__ u16  sA [64 * AP];        // Xs tile       [m][k]    66,560 B
  __shared__ u16  sB1[2][64 * AP];     // W1T chunk     [ff][k]  133,120 B (double-buffered)
  __shared__ u16  sH [64 * HP];        // relu(H) chunk [m][ff]    9,216 B
  __shared__ u16  sB2[512 * HP];       // W2T chunk     [d][ff]   73,728 B
  __shared__ float sb1[FFDIM];         //                           8,192 B
  __shared__ float sb2[DIM];           //                           2,048 B

  int4 ti = tiles[blockIdx.x];
  int e = ti.x, row0 = ti.y, mrows = ti.z;
  if (mrows == 0) return;

  int tid = threadIdx.x, lane = tid & 31, wave = tid >> 5;
  int mt = wave & 3;                // this wave's 16-row M block
  int nh = wave >> 2;               // 0/1: N-half (GEMM2) / FF-half (GEMM1)

  const u16* gA  = xs_sorted + (size_t)row0 * DIM;
  const u16* gB1 = W1T + (size_t)e * FFDIM * DIM;
  const u16* gB2 = W2T + (size_t)e * DIM * FFDIM;

  // ---- prologue: async-issue sA, sB1[0](chunk0), sB2(chunk0) -- strictly this order
  #pragma unroll
  for (int t2 = 0; t2 < 16; ++t2) {              // Xs tile: 64 rows x 64 x 16B
    int idx = tid + t2 * 256;
    int r = idx >> 6, c = idx & 63;
    if (r < mrows)
      async_ld128(&sA[r * AP + c * 8], gA, (unsigned)(r * 64 + c) * 16u);
  }
  #pragma unroll
  for (int t2 = 0; t2 < 16; ++t2) {              // W1T rows [0,64) all K
    int idx = tid + t2 * 256;
    int r = idx >> 6, c = idx & 63;
    async_ld128(&sB1[0][r * AP + c * 8], gB1, (unsigned)(r * 64 + c) * 16u);
  }
  #pragma unroll
  for (int t2 = 0; t2 < 16; ++t2) {              // W2T all d, ff cols [0,64)
    int idx = tid + t2 * 256;
    int r = idx >> 3, c = idx & 7;
    async_ld128(&sB2[r * HP + c * 8], gB2, (unsigned)(r * 256 + c) * 16u);
  }
  // zero-fill tail rows of sA (ds stores; disjoint from async-load targets)
  if (mrows < 64) {
    const uint4 z = make_uint4(0, 0, 0, 0);
    for (int idx = tid; idx < (64 - mrows) * 64; idx += 256) {
      int r = mrows + (idx >> 6), c = idx & 63;
      *(uint4*)&sA[r * AP + c * 8] = z;
    }
  }
  for (int i = tid; i < FFDIM; i += 256) sb1[i] = b1[(size_t)e * FFDIM + i];
  for (int i = tid; i < DIM;   i += 256) sb2[i] = b2[(size_t)e * DIM + i];

  v8f acc[16];
  #pragma unroll
  for (int i = 0; i < 16; ++i) acc[i] = (v8f)0.f;

  int buf = 0;
  for (int chunk = 0; chunk < 32; ++chunk) {
    int nc = (chunk + 1) & 31;                   // wrap keeps wait thresholds uniform

    // sB1[buf] (and sA) ready: newest 16 outstanding are this chunk's sB2
    WAIT_ASYNC_16();
    __syncthreads();

    // ---- Stage A: Hc[64,64] = sA @ W1chunk ; M block mt, FF tiles {nh*2, nh*2+1}
    v8f h0 = (v8f)0.f, h1 = (v8f)0.f;
    const u16* sB1c = sB1[buf];
    #pragma unroll
    for (int ks = 0; ks < 16; ++ks) {
      int kb = ks * 32;
      v16bf a  = frag_load(sA,   mt * 16,      AP, kb, lane);
      v16bf bA = frag_load(sB1c, nh * 32,      AP, kb, lane);
      v16bf bB = frag_load(sB1c, nh * 32 + 16, AP, kb, lane);
      h0 = __builtin_amdgcn_wmma_f32_16x16x32_bf16(false, a, false, bA, (short)0, h0, false, false);
      h1 = __builtin_amdgcn_wmma_f32_16x16x32_bf16(false, a, false, bB, (short)0, h1, false, false);
    }

    // issue next W1T chunk into the idle buffer (overlaps Stage B)
    #pragma unroll
    for (int t2 = 0; t2 < 16; ++t2) {
      int idx = tid + t2 * 256;
      int r = idx >> 6, c = idx & 63;
      async_ld128(&sB1[buf ^ 1][r * AP + c * 8], gB1, (unsigned)((nc * 64 + r) * 64 + c) * 16u);
    }

    // bias + relu + pack bf16 into sH
    {
      int nIdx = lane & 15, half = lane >> 4;
      int f0 = nh * 32 + nIdx, f1 = f0 + 16;
      float bia0 = sb1[chunk * 64 + f0], bia1 = sb1[chunk * 64 + f1];
      #pragma unroll
      for (int rI = 0; rI < 8; ++rI) {
        int m = mt * 16 + 8 * half + rI;
        sH[m * HP + f0] = f32_to_bf16(fmaxf(h0[rI] + bia0, 0.f));
        sH[m * HP + f1] = f32_to_bf16(fmaxf(h1[rI] + bia1, 0.f));
      }
    }

    // sB2(chunk) ready: newest 16 outstanding are sB1[next]; barrier also publishes sH
    WAIT_ASYNC_16();
    __syncthreads();

    // ---- Stage B: Y[64,512] += Hc @ W2chunk ; M block mt, N tiles [nh*16, +16)
    #pragma unroll
    for (int ks = 0; ks < 2; ++ks) {
      int kb = ks * 32;
      v16bf a = frag_load(sH, mt * 16, HP, kb, lane);
      #pragma unroll
      for (int nt = 0; nt < 16; ++nt) {
        v16bf b = frag_load(sB2, (nh * 16 + nt) * 16, HP, kb, lane);
        acc[nt] = __builtin_amdgcn_wmma_f32_16x16x32_bf16(false, a, false, b, (short)0, acc[nt], false, false);
      }
    }
    __syncthreads();   // all waves done reading sB2

    // issue next W2T chunk (overlaps next chunk's Stage A)
    #pragma unroll
    for (int t2 = 0; t2 < 16; ++t2) {
      int idx = tid + t2 * 256;
      int r = idx >> 3, c = idx & 7;
      async_ld128(&sB2[r * HP + c * 8], gB2, (unsigned)(r * 256 + nc * 8 + c) * 16u);
    }
    buf ^= 1;
  }

  // ---- epilogue: + b2, store f32 rows directly at sorted output positions
  {
    int nIdx = lane & 15, half = lane >> 4;
    #pragma unroll
    for (int nt = 0; nt < 16; ++nt) {
      int n = (nh * 16 + nt) * 16 + nIdx;
      float bia = sb2[n];
      #pragma unroll
      for (int rI = 0; rI < 8; ++rI) {
        int m = mt * 16 + 8 * half + rI;
        if (m < mrows) out[(size_t)(row0 + m) * DIM + n] = acc[nt][rI] + bia;
      }
    }
  }
}

// ---------------- host launcher ----------------
extern "C" void kernel_launch(void* const* d_in, const int* in_sizes, int n_in,
                              void* d_out, int out_size, void* d_ws, size_t ws_size,
                              hipStream_t stream) {
  (void)in_sizes; (void)n_in; (void)out_size; (void)ws_size;
  const float* x  = (const float*)d_in[0];
  const float* Wg = (const float*)d_in[1];
  const float* bg = (const float*)d_in[2];
  const float* W1 = (const float*)d_in[3];
  const float* b1 = (const float*)d_in[4];
  const float* W2 = (const float*)d_in[5];
  const float* b2 = (const float*)d_in[6];
  float* out = (float*)d_out;

  char* w = (char*)d_ws;
  const size_t SZ_W = (size_t)NEXP * DIM * FFDIM * 2;   // 16 MB each
  u16*   W1T        = (u16*)(w);
  u16*   W2T        = (u16*)(w + SZ_W);
  u16*   xs_sorted  = (u16*)(w + 2 * SZ_W);
  int*   route      = (int*)(w + 3 * SZ_W);
  float* pmaxv      = (float*)(w + 3 * SZ_W + (size_t)T_TOK * 4);
  int*   blockCnt   = (int*)(w + 3 * SZ_W + (size_t)T_TOK * 8);
  int*   baseArr    = (int*)(w + 3 * SZ_W + (size_t)T_TOK * 8 + 2048);
  int4*  tiles      = (int4*)(w + 3 * SZ_W + (size_t)T_TOK * 8 + 4096);

  // W1 [E][D][FF] -> W1T [E][FF][D]
  transpose_bf16_kernel<<<dim3(FFDIM/32, DIM/32, NEXP), dim3(32,8), 0, stream>>>(W1, W1T, DIM, FFDIM);
  // W2 [E][FF][D] -> W2T [E][D][FF]
  transpose_bf16_kernel<<<dim3(DIM/32, FFDIM/32, NEXP), dim3(32,8), 0, stream>>>(W2, W2T, FFDIM, DIM);

  router_kernel<<<T_TOK/8, 256, 0, stream>>>(x, Wg, bg, route, pmaxv);
  hist_kernel<<<T_TOK/256, 256, 0, stream>>>(route, blockCnt);
  scan_kernel<<<1, 32, 0, stream>>>(blockCnt, baseArr, tiles);
  gather_kernel<<<T_TOK/256, 256, 0, stream>>>(x, route, pmaxv, baseArr, xs_sorted);
  ffn_kernel<<<MAXTILES, 256, 0, stream>>>(xs_sorted, W1T, W2T, b1, b2, tiles, out);
}
